// ProjectionDecoder_31817117729451
// MI455X (gfx1250) — matrix operand order repaired and checked by reading, not compiled, not run on hardware
//
#include <hip/hip_runtime.h>
#include <hip/hip_bf16.h>
#include <math.h>

// ---------------------------------------------------------------------------
// ProjectionDecoder pipeline for gfx1250 (MI455X, wave32, WMMA)
//   angles(2048,3) f32, spectrum(49,10,2) f32, wt(10,1,4,4) f32, bt(1) f32
//   out: (2048,1,128,128) f32
// Stage 2 (spherical inverse transform) is a GEMM done with
// v_wmma_f32_16x16x32_f16; stages 3+4 are fused to avoid materializing the
// 335MB projection tensor (bandwidth is the only thing that matters here).
// ---------------------------------------------------------------------------

typedef __attribute__((ext_vector_type(16))) _Float16 v16h;
typedef __attribute__((ext_vector_type(8)))  _Float16 v8h;
typedef __attribute__((ext_vector_type(8)))  float    v8f;

#define PI_F 3.14159265358979323846f

// GEMM dims: rows = n*rep = 20480, K = 98 -> padded 128, N = 196 -> padded 208
#define GM   20480
#define GK   128
#define GNP  208
#define GN   196

// workspace layout (bytes)
#define WS_OFF_A   0u                            // 20480*128 halfs = 5,242,880
#define WS_OFF_B   5242880u                      // 208*128 halfs   =    53,248
#define WS_OFF_SIG 5296128u                      // 20480*196 f32   = 16,056,320

static __device__ __forceinline__ float powi_f(float x, int e) {
    float r = 1.0f;
    for (int i = 0; i < e; ++i) r *= x;
    return r;
}

// ---------------------------------------------------------------------------
// Kernel 1: build packed fp16 B matrix, Bt[ncol][k]  (ncol-major, K contiguous)
//   rows k in [0,49)  : Yre[k, j, kk]
//   rows k in [49,98) : -Yim[k-49, j, kk]
//   rows k in [98,128): 0  ; cols >= 196: 0
// ---------------------------------------------------------------------------
__global__ void k_build_B(_Float16* __restrict__ Bt) {
    const int t = threadIdx.x;           // 256 threads, 1 block
    if (t >= GNP) return;
    _Float16* col = Bt + (size_t)t * GK;
    for (int k = 0; k < GK; ++k) col[k] = (_Float16)0.0f;
    if (t >= GN) return;

    const int j  = t / 14;               // beta index
    const int kk = t % 14;               // alpha index
    const float beta  = PI_F * (float)(2 * j + 1) / 28.0f;   // pi(2j+1)/(4B), B=7
    const float alpha = PI_F * (float)kk / 7.0f;             // 2*pi*kk/14
    const float x  = cosf(beta);
    const float sx = sinf(beta);

    float FACT[13];
    FACT[0] = 1.0f;
    #pragma unroll
    for (int i = 1; i < 13; ++i) FACT[i] = FACT[i - 1] * (float)i;

    float P[7][7];
    #pragma unroll
    for (int a = 0; a < 7; ++a)
        #pragma unroll
        for (int b = 0; b < 7; ++b) P[a][b] = 0.0f;
    P[0][0] = 1.0f;
    #pragma unroll
    for (int m = 1; m < 7; ++m) P[m][m] = -(float)(2 * m - 1) * sx * P[m - 1][m - 1];
    #pragma unroll
    for (int m = 0; m < 7; ++m) {
        if (m + 1 <= 6) P[m + 1][m] = (float)(2 * m + 1) * x * P[m][m];
        #pragma unroll
        for (int l = 0; l <= 6; ++l) {
            if (l >= m + 2)
                P[l][m] = ((float)(2 * l - 1) * x * P[l - 1][m]
                           - (float)(l + m - 1) * P[l - 2][m]) / (float)(l - m);
        }
    }

    #pragma unroll
    for (int l = 0; l <= 6; ++l) {
        #pragma unroll
        for (int m = 0; m <= 6; ++m) {
            if (m > l) continue;
            const float Nn  = sqrtf((float)(2 * l + 1) / (4.0f * PI_F) * FACT[l - m] / FACT[l + m]);
            const float pr  = Nn * P[l][m];
            const float yre = pr * cosf((float)m * alpha);
            const float yim = pr * sinf((float)m * alpha);
            col[l * l + l + m]      = (_Float16)yre;
            col[49 + l * l + l + m] = (_Float16)(-yim);
            if (m > 0) {
                const float sgn = (m & 1) ? -1.0f : 1.0f;
                // Yre[-m] = sgn*yre ; Yim[-m] = -sgn*yim  -> stored -Yim = sgn*yim
                col[l * l + l - m]      = (_Float16)(sgn * yre);
                col[49 + l * l + l - m] = (_Float16)(sgn * yim);
            }
        }
    }
}

// ---------------------------------------------------------------------------
// Kernel 2: Wigner-D rotation of the spectrum -> fp16 A matrix
//   A[row = n*10 + r][k] : k<49 -> Sre[n,k,r], 49<=k<98 -> Sim[n,k-49,r], pad 0
//   one block (64 threads) per sample; thread t<49 owns output row (l,m')
// ---------------------------------------------------------------------------
__global__ void k_rotate(const float* __restrict__ angles,
                         const float* __restrict__ spectrum,
                         _Float16* __restrict__ A) {
    const int n = blockIdx.x;            // 2048
    const int t = threadIdx.x;           // 64
    const float alpha = angles[n * 3 + 0];
    const float beta  = angles[n * 3 + 1];
    const float gamma = angles[n * 3 + 2];
    const float ch = cosf(beta * 0.5f);
    const float sh = sinf(beta * 0.5f);

    // zero the K padding (cols 98..127) of this sample's 10 rows
    for (int idx = t; idx < 10 * 30; idx += 64) {
        const int r = idx / 30, c = 98 + idx % 30;
        A[(size_t)(n * 10 + r) * GK + c] = (_Float16)0.0f;
    }
    if (t >= 49) return;

    int l = 0;
    while ((l + 1) * (l + 1) <= t) ++l;
    const int mp = t - l * l - l;

    float FACT[13];
    FACT[0] = 1.0f;
    #pragma unroll
    for (int i = 1; i < 13; ++i) FACT[i] = FACT[i - 1] * (float)i;

    float accRe[10], accIm[10];
    #pragma unroll
    for (int r = 0; r < 10; ++r) { accRe[r] = 0.0f; accIm[r] = 0.0f; }

    for (int j = 0; j <= 2 * l; ++j) {
        const int m = j - l;
        const float pref = sqrtf(FACT[l + mp] * FACT[l - mp] * FACT[l + m] * FACT[l - m]);
        const int k0 = (m - mp) > 0 ? (m - mp) : 0;
        const int k1 = (l + m) < (l - mp) ? (l + m) : (l - mp);
        float d = 0.0f;
        for (int k = k0; k <= k1; ++k) {
            float c = pref / (FACT[l + m - k] * FACT[k] * FACT[l - mp - k] * FACT[mp - m + k]);
            if ((mp - m + k) & 1) c = -c;
            d += c * powi_f(ch, 2 * l + m - mp - 2 * k) * powi_f(sh, mp - m + 2 * k);
        }
        const float ang = (float)mp * alpha + (float)m * gamma;
        const float Dre =  d * cosf(ang);
        const float Dim = -d * sinf(ang);
        const float* sp = spectrum + (size_t)(l * l + j) * 20;   // (m, rep, 2)
        #pragma unroll
        for (int r = 0; r < 10; ++r) {
            const float sre = sp[r * 2 + 0];
            const float sim = sp[r * 2 + 1];
            accRe[r] += Dre * sre - Dim * sim;
            accIm[r] += Dre * sim + Dim * sre;
        }
    }
    #pragma unroll
    for (int r = 0; r < 10; ++r) {
        _Float16* row = A + (size_t)(n * 10 + r) * GK;
        row[t]      = (_Float16)accRe[r];
        row[49 + t] = (_Float16)accIm[r];
    }
}

// ---------------------------------------------------------------------------
// Kernel 3: WMMA GEMM  Sig[M=20480, 196] = A[M,128] * Bt^T  (fp16 in, f32 acc)
//   256 threads = 8 waves; each wave owns a 16-row strip x all 13 N tiles.
// ---------------------------------------------------------------------------
__global__ void k_gemm(const _Float16* __restrict__ A,
                       const _Float16* __restrict__ Bt,
                       float* __restrict__ Sig) {
    const int wave    = threadIdx.x >> 5;    // wave32
    const int lane    = threadIdx.x & 31;
    const int rowbase = blockIdx.x * 128 + wave * 16;
    const int hsel    = lane >> 4;           // 0: lanes 0-15, 1: lanes 16-31
    const int rc      = lane & 15;           // A row / B col within tile

    v8f acc[13];
    #pragma unroll
    for (int nt = 0; nt < 13; ++nt)
        acc[nt] = (v8f){0.f, 0.f, 0.f, 0.f, 0.f, 0.f, 0.f, 0.f};

    const _Float16* arow = A + (size_t)(rowbase + rc) * GK;

    #pragma unroll
    for (int ks = 0; ks < 4; ++ks) {
        const int k0 = ks * 32;
        // A fragment (16x32 f16): lanes 0-15 hold K=[0..7,16..23] of this step,
        // lanes 16-31 hold K=[8..15,24..31]; row = lane&15.
        const _Float16* pa = arow + k0 + hsel * 8;
        const v8h alo = *(const v8h*)(pa);
        const v8h ahi = *(const v8h*)(pa + 16);
        const v16h af = __builtin_shufflevector(alo, ahi,
            0, 1, 2, 3, 4, 5, 6, 7, 8, 9, 10, 11, 12, 13, 14, 15);
        if (ks < 3) __builtin_prefetch(arow + k0 + 32, 0, 3);   // global_prefetch_b8

        #pragma unroll
        for (int nt = 0; nt < 13; ++nt) {
            // B fragment (32x16 f16): col = lane&15; lanes 0-15 K=[0..7,16..23],
            // lanes 16-31 K=[8..15,24..31] (Bt is col-major over N, K fast).
            const _Float16* pb = Bt + (size_t)(nt * 16 + rc) * GK + k0 + hsel * 8;
            const v8h blo = *(const v8h*)(pb);
            const v8h bhi = *(const v8h*)(pb + 16);
            const v16h bf = __builtin_shufflevector(blo, bhi,
                0, 1, 2, 3, 4, 5, 6, 7, 8, 9, 10, 11, 12, 13, 14, 15);
            acc[nt] = __builtin_amdgcn_wmma_f32_16x16x32_f16(
                false, af, false, bf, (short)0, acc[nt], false, false);
        }
    }

    // C/D layout: VGPR v -> M = v + 8*hsel, N = lane&15 within tile
    #pragma unroll
    for (int nt = 0; nt < 13; ++nt) {
        const int col = nt * 16 + rc;
        if (col < GN) {
            #pragma unroll
            for (int v = 0; v < 8; ++v) {
                const int mrow = v + hsel * 8;
                Sig[(size_t)(rowbase + mrow) * GN + col] = acc[nt][v];
            }
        }
    }
}

// ---------------------------------------------------------------------------
// Kernel 4: fused orthographic grid-sample (14x14 -> 64x64 bilinear) +
//           ConvTranspose2d(10->1, k=4, s=2, p=1). One block per sample.
//           Signal (10x14x14) and weights live in LDS; the 335MB projection
//           tensor is never materialized.
// ---------------------------------------------------------------------------
__global__ void k_project_deconv(const float* __restrict__ Sig,
                                 const float* __restrict__ wt,
                                 const float* __restrict__ bt,
                                 float* __restrict__ out) {
    __shared__ float s_sig[10 * 196];
    __shared__ float s_wt[160];
    const int n = blockIdx.x;            // 2048
    const int t = threadIdx.x;           // 256

    for (int idx = t; idx < 1960; idx += 256) s_sig[idx] = Sig[(size_t)n * 1960 + idx];
    for (int idx = t; idx < 160;  idx += 256) s_wt[idx]  = wt[idx];
    __syncthreads();
    const float bias = bt[0];

    for (int p = t; p < 128 * 128; p += 256) {
        const int oy = p >> 7;
        const int ox = p & 127;
        float acc = bias;
        const int py = oy & 1;
        const int px = ox & 1;
        #pragma unroll
        for (int ty = 0; ty < 2; ++ty) {
            const int ky = py + 2 * ty;
            const int dy = oy + ky - 2;      // dilated row index, always even
            const int iy = dy >> 1;
            if (dy < 0 || iy >= 64) continue;
            #pragma unroll
            for (int tx = 0; tx < 2; ++tx) {
                const int kx = px + 2 * tx;
                const int dx = ox + kx - 2;
                const int ix = dx >> 1;
                if (dx < 0 || ix >= 64) continue;

                // orthographic grid point (iy, ix) -> sample coords in 14x14
                const float ys   = -0.8f + (float)iy * (1.6f / 63.0f);
                const float xs   = -0.8f + (float)ix * (1.6f / 63.0f);
                const float rho  = sqrtf(xs * xs + ys * ys);
                const float th   = asinf(fminf(rho, 1.0f));
                const float phi  = atan2f(ys, xs);
                const float gx   = phi * (1.0f / PI_F);
                const float gy   = th * (2.0f / PI_F) - 1.0f;
                const float fx   = (gx + 1.0f) * 7.0f - 0.5f;
                const float fy   = (gy + 1.0f) * 7.0f - 0.5f;
                const float x0f  = floorf(fx);
                const float y0f  = floorf(fy);
                const int   x0   = (int)x0f;
                const int   y0   = (int)y0f;
                const float wx1  = fx - x0f, wx0 = 1.0f - wx1;
                const float wy1  = fy - y0f, wy0 = 1.0f - wy1;
                const float vx0  = (x0 >= 0 && x0 < 14)       ? 1.0f : 0.0f;
                const float vx1  = (x0 + 1 >= 0 && x0 + 1 < 14) ? 1.0f : 0.0f;
                const float vy0  = (y0 >= 0 && y0 < 14)       ? 1.0f : 0.0f;
                const float vy1  = (y0 + 1 >= 0 && y0 + 1 < 14) ? 1.0f : 0.0f;
                const float w00 = wx0 * wy0 * vx0 * vy0;
                const float w10 = wx1 * wy0 * vx1 * vy0;
                const float w01 = wx0 * wy1 * vx0 * vy1;
                const float w11 = wx1 * wy1 * vx1 * vy1;
                const int xc0 = x0 < 0 ? 0 : (x0 > 13 ? 13 : x0);
                const int xc1 = x0 + 1 < 0 ? 0 : (x0 + 1 > 13 ? 13 : x0 + 1);
                const int yc0 = y0 < 0 ? 0 : (y0 > 13 ? 13 : y0);
                const int yc1 = y0 + 1 < 0 ? 0 : (y0 + 1 > 13 ? 13 : y0 + 1);
                const int i00 = yc0 * 14 + xc0;
                const int i10 = yc0 * 14 + xc1;
                const int i01 = yc1 * 14 + xc0;
                const int i11 = yc1 * 14 + xc1;
                const int kidx = (3 - ky) * 4 + (3 - kx);
                #pragma unroll
                for (int cin = 0; cin < 10; ++cin) {
                    const float* sc = s_sig + cin * 196;
                    const float bil = w00 * sc[i00] + w10 * sc[i10]
                                    + w01 * sc[i01] + w11 * sc[i11];
                    acc += s_wt[cin * 16 + kidx] * bil;
                }
            }
        }
        out[(size_t)n * 16384 + p] = acc;
    }
}

// ---------------------------------------------------------------------------
extern "C" void kernel_launch(void* const* d_in, const int* in_sizes, int n_in,
                              void* d_out, int out_size, void* d_ws, size_t ws_size,
                              hipStream_t stream) {
    const float* angles   = (const float*)d_in[0];   // (2048,3)
    const float* spectrum = (const float*)d_in[1];   // (49,10,2)
    const float* wt       = (const float*)d_in[2];   // (10,1,4,4)
    const float* bt       = (const float*)d_in[3];   // (1,)
    float* out = (float*)d_out;                      // (2048,1,128,128)

    char* ws = (char*)d_ws;
    _Float16* A   = (_Float16*)(ws + WS_OFF_A);      // 20480 x 128 f16
    _Float16* Bt  = (_Float16*)(ws + WS_OFF_B);      // 208 x 128 f16 (N-major)
    float*    Sig = (float*)(ws + WS_OFF_SIG);       // 20480 x 196 f32

    k_build_B<<<1, 256, 0, stream>>>(Bt);
    k_rotate<<<2048, 64, 0, stream>>>(angles, spectrum, A);
    k_gemm<<<GM / 128, 256, 0, stream>>>(A, Bt, Sig);
    k_project_deconv<<<2048, 256, 0, stream>>>(Sig, wt, bt, out);
}
